// FastNGPNeRF_68298569941738
// MI455X (gfx1250) — matrix operand order repaired and checked by reading, not compiled, not run on hardware
//
#include <hip/hip_runtime.h>
#include <hip/hip_bf16.h>
#include <math.h>

typedef __attribute__((ext_vector_type(16))) _Float16 v16h;
typedef __attribute__((ext_vector_type(8)))  float    v8f;

#define NLVL   16
#define LOG2T  19
#define TSZ    (1u << LOG2T)
#define TMASK  (TSZ - 1u)
#define PRIME1 2654435761u
#define PRIME2 805459861u

__constant__ float RES_TAB[NLVL] = {
  16.f, 20.f, 25.f, 32.f, 40.f, 50.f, 64.f, 80.f,
  101.f, 128.f, 161.f, 203.f, 256.f, 322.f, 406.f, 512.f
};

// A-fragment (16x32 f16 layout) from f32 LDS activations, row stride 64.
// lane half: K = kchunk*32 + {half*8+j (j<8), 16+half*8+(j-8)}
static __device__ inline v16h load_a_f32(const float* __restrict__ buf, int m, int hlf, int kchunk) {
  const float* r = buf + m * 64 + kchunk * 32;
  v16h a;
#pragma unroll
  for (int j = 0; j < 8; ++j) a[j] = (_Float16)r[hlf * 8 + j];
#pragma unroll
  for (int j = 0; j < 8; ++j) a[8 + j] = (_Float16)r[16 + hlf * 8 + j];
  return a;
}

// B-fragment (KxN f16) from weights stored row-major [N][K] (i.e. B = W^T).
// b[j] = W[n][kchunk*32 + half*8 + j], b[8+j] = W[n][kchunk*32 + 16 + half*8 + j].
static __device__ inline v16h load_b_f16(const _Float16* __restrict__ w, int K, int n, int hlf, int kchunk) {
  const _Float16* r = w + n * K + kchunk * 32 + hlf * 8;
  v16h b;
#pragma unroll
  for (int j = 0; j < 16; ++j) b[j] = r[(j < 8) ? j : (8 + j)];
  return b;
}

__global__ __launch_bounds__(128) void ngp_fused_kernel(
    const float* __restrict__ x, const float* __restrict__ dirs,
    const float* __restrict__ table,
    const float* __restrict__ W1, const float* __restrict__ B1,
    const float* __restrict__ W2, const float* __restrict__ B2,
    const float* __restrict__ C1, const float* __restrict__ C1b,
    const float* __restrict__ C2, const float* __restrict__ C2b,
    float* __restrict__ out_rgb, float* __restrict__ out_sigma, int npts)
{
  __shared__ __align__(16) _Float16 sW1[64 * 32];   // [n][k] = W1 as given
  __shared__ __align__(16) _Float16 sW2[16 * 64];
  __shared__ __align__(16) _Float16 sC1[64 * 64];   // K padded 42->64 with zeros
  __shared__ __align__(16) _Float16 sC2[16 * 64];   // N padded 3->16 with zero rows
  __shared__ float sB1[64], sB2[16], sC1bs[64], sC2bs[4];
  __shared__ __align__(16) float bufH[4][16 * 64];  // per-wave: xenc -> h -> h2
  __shared__ __align__(16) float bufC[4][16 * 64];  // per-wave: ci (feats|direnc|pad)

  const int tid = threadIdx.x;

  // ---- stage weights into LDS as f16 ----
  for (int i = tid; i < 64 * 32; i += 128) sW1[i] = (_Float16)W1[i];
  for (int i = tid; i < 16 * 64; i += 128) sW2[i] = (_Float16)W2[i];
  for (int i = tid; i < 64 * 64; i += 128) {
    int n = i >> 6, k = i & 63;
    sC1[i] = (_Float16)((k < 42) ? C1[n * 42 + k] : 0.f);
  }
  for (int i = tid; i < 16 * 64; i += 128) {
    int n = i >> 6, k = i & 63;
    sC2[i] = (_Float16)((n < 3) ? C2[n * 64 + k] : 0.f);
  }
  if (tid < 64) { sB1[tid] = B1[tid]; sC1bs[tid] = C1b[tid]; }
  if (tid < 16) sB2[tid] = B2[tid];
  if (tid < 4)  sC2bs[tid] = (tid < 3) ? C2b[tid] : 0.f;
  __syncthreads();

  const int wave = tid >> 5;
  const int lane = tid & 31;
  const int hlf  = lane >> 4;   // 0/1
  const int mn   = lane & 15;   // row (M) or col (N) index depending on role
  const int base = (blockIdx.x * 4 + wave) * 16;
  float* __restrict__ myH = bufH[wave];
  float* __restrict__ myC = bufC[wave];

  // ---- phase 1a: hash-grid encode: lane does 8 levels of point mn ----
  {
    int p = base + mn; if (p >= npts) p = npts - 1;
    float px = x[p * 3 + 0], py = x[p * 3 + 1], pz = x[p * 3 + 2];
#pragma unroll
    for (int i = 0; i < 8; ++i) {
      int l = hlf * 8 + i;
      float r = RES_TAB[l];
      float sx = px * r, sy = py * r, sz = pz * r;
      float fx = floorf(sx), fy = floorf(sy), fz = floorf(sz);
      float tx = sx - fx, ty = sy - fy, tz = sz - fz;
      unsigned ix = (unsigned)fx, iy = (unsigned)fy, iz = (unsigned)fz;
      unsigned hy0 = iy * PRIME1, hy1 = hy0 + PRIME1;
      unsigned hz0 = iz * PRIME2, hz1 = hz0 + PRIME2;
      const float2* tl = (const float2*)table + (size_t)l * TSZ;
      float f0 = 0.f, f1 = 0.f;
#pragma unroll
      for (int c = 0; c < 8; ++c) {
        unsigned cx = ix + ((c >> 2) & 1);
        unsigned hh = (cx ^ ((c & 2) ? hy1 : hy0) ^ ((c & 1) ? hz1 : hz0)) & TMASK;
        float2 f = tl[hh];
        float w = ((c & 4) ? tx : 1.f - tx) * ((c & 2) ? ty : 1.f - ty) * ((c & 1) ? tz : 1.f - tz);
        f0 += w * f.x; f1 += w * f.y;
      }
      myH[mn * 64 + 2 * l + 0] = f0;
      myH[mn * 64 + 2 * l + 1] = f1;
    }
  }
  // ---- phase 1b: dir encode into ci cols 15..41; zero pad cols 42..63 ----
  if (hlf == 0) {
    int p = base + mn; if (p >= npts) p = npts - 1;
    float dx = dirs[p * 3], dy = dirs[p * 3 + 1], dz = dirs[p * 3 + 2];
    float inv = rsqrtf(dx * dx + dy * dy + dz * dz);
    dx *= inv; dy *= inv; dz *= inv;
    float* row = myC + mn * 64;
    row[15] = dx; row[16] = dy; row[17] = dz;
    float dd[3] = {dx, dy, dz};
#pragma unroll
    for (int di = 0; di < 3; ++di) {
#pragma unroll
      for (int fq = 0; fq < 4; ++fq) {
        float ang = dd[di] * (float)(1 << fq);
        row[18 + di * 4 + fq] = __sinf(ang);
        row[30 + di * 4 + fq] = __cosf(ang);
      }
    }
  } else {
    float* row = myC + mn * 64;
#pragma unroll
    for (int k = 42; k < 64; ++k) row[k] = 0.f;
  }
  __syncthreads();

  // ---- density layer 1: xenc[16x32] @ W1^T[32x64] + b1, relu -> myH[16][64] ----
  v16h a1 = load_a_f32(myH, mn, hlf, 0);
  v8f acc[4];
#pragma unroll
  for (int nt = 0; nt < 4; ++nt) {
    v16h b = load_b_f16(sW1, 32, nt * 16 + mn, hlf, 0);
    v8f cz = {};
    acc[nt] = __builtin_amdgcn_wmma_f32_16x16x32_f16(false, a1, false, b, (short)0, cz, false, false);
  }
  __syncthreads();
#pragma unroll
  for (int nt = 0; nt < 4; ++nt) {
    float bb = sB1[nt * 16 + mn];
#pragma unroll
    for (int i = 0; i < 8; ++i)
      myH[(hlf * 8 + i) * 64 + nt * 16 + mn] = fmaxf(acc[nt][i] + bb, 0.f);
  }
  __syncthreads();

  // ---- density layer 2: h[16x64] @ W2^T[64x16] + b2 -> sigma + feats ----
  {
    v16h a20 = load_a_f32(myH, mn, hlf, 0);
    v16h a21 = load_a_f32(myH, mn, hlf, 1);
    v16h b20 = load_b_f16(sW2, 64, mn, hlf, 0);
    v16h b21 = load_b_f16(sW2, 64, mn, hlf, 1);
    v8f df = {};
    df = __builtin_amdgcn_wmma_f32_16x16x32_f16(false, a20, false, b20, (short)0, df, false, false);
    df = __builtin_amdgcn_wmma_f32_16x16x32_f16(false, a21, false, b21, (short)0, df, false, false);
    float bb = sB2[mn];
    if (mn == 0) {
#pragma unroll
      for (int i = 0; i < 8; ++i) {
        int row = hlf * 8 + i;
        if (base + row < npts) out_sigma[base + row] = fmaxf(df[i] + bb, 0.f);
      }
    } else {
#pragma unroll
      for (int i = 0; i < 8; ++i)
        myC[(hlf * 8 + i) * 64 + (mn - 1)] = df[i] + bb;   // feats -> ci cols 0..14
    }
  }
  __syncthreads();

  // ---- color layer 1: ci[16x64(42)] @ C1^T + c1, relu -> myH ----
  {
    v16h ca0 = load_a_f32(myC, mn, hlf, 0);
    v16h ca1 = load_a_f32(myC, mn, hlf, 1);
    v8f cacc[4];
#pragma unroll
    for (int nt = 0; nt < 4; ++nt) {
      v8f cz = {};
      cz = __builtin_amdgcn_wmma_f32_16x16x32_f16(false, ca0, false,
             load_b_f16(sC1, 64, nt * 16 + mn, hlf, 0), (short)0, cz, false, false);
      cz = __builtin_amdgcn_wmma_f32_16x16x32_f16(false, ca1, false,
             load_b_f16(sC1, 64, nt * 16 + mn, hlf, 1), (short)0, cz, false, false);
      cacc[nt] = cz;
    }
    __syncthreads();
#pragma unroll
    for (int nt = 0; nt < 4; ++nt) {
      float bb = sC1bs[nt * 16 + mn];
#pragma unroll
      for (int i = 0; i < 8; ++i)
        myH[(hlf * 8 + i) * 64 + nt * 16 + mn] = fmaxf(cacc[nt][i] + bb, 0.f);
    }
    __syncthreads();
  }

  // ---- color layer 2: h2[16x64] @ C2^T[64x16(3)] + c2, sigmoid -> rgb ----
  {
    v16h fa0 = load_a_f32(myH, mn, hlf, 0);
    v16h fa1 = load_a_f32(myH, mn, hlf, 1);
    v16h fb0 = load_b_f16(sC2, 64, mn, hlf, 0);
    v16h fb1 = load_b_f16(sC2, 64, mn, hlf, 1);
    v8f o = {};
    o = __builtin_amdgcn_wmma_f32_16x16x32_f16(false, fa0, false, fb0, (short)0, o, false, false);
    o = __builtin_amdgcn_wmma_f32_16x16x32_f16(false, fa1, false, fb1, (short)0, o, false, false);
    if (mn < 3) {
      float bb = sC2bs[mn];
#pragma unroll
      for (int i = 0; i < 8; ++i) {
        int row = hlf * 8 + i;
        if (base + row < npts) {
          float v = o[i] + bb;
          out_rgb[(base + row) * 3 + mn] = 1.f / (1.f + __expf(-v));
        }
      }
    }
  }
}

extern "C" void kernel_launch(void* const* d_in, const int* in_sizes, int n_in,
                              void* d_out, int out_size, void* d_ws, size_t ws_size,
                              hipStream_t stream) {
  const float* x     = (const float*)d_in[0];
  const float* dirs  = (const float*)d_in[1];
  const float* table = (const float*)d_in[2];
  const float* W1    = (const float*)d_in[3];
  const float* B1    = (const float*)d_in[4];
  const float* W2    = (const float*)d_in[5];
  const float* B2    = (const float*)d_in[6];
  const float* C1    = (const float*)d_in[7];
  const float* C1b   = (const float*)d_in[8];
  const float* C2    = (const float*)d_in[9];
  const float* C2b   = (const float*)d_in[10];

  const int npts = in_sizes[0] / 3;
  float* out_rgb   = (float*)d_out;                       // [N,3]
  float* out_sigma = (float*)d_out + (size_t)3 * npts;    // [N]

  const int nblk = (npts + 63) / 64;   // 4 waves/block * 16 pts/wave
  ngp_fused_kernel<<<nblk, 128, 0, stream>>>(
      x, dirs, table, W1, B1, W2, B2, C1, C1b, C2, C2b,
      out_rgb, out_sigma, npts);
}
